// GlobalModel_57088705298519
// MI455X (gfx1250) — matrix operand order repaired and checked by reading, not compiled, not run on hardware
//
#include <hip/hip_runtime.h>
#include <hip/hip_bf16.h>

#define BQ  16
#define LQ  24
#define NQ  256
#define EQ  4096
#define EMQ 16
#define HQ  64
#define GQ  64
#define TQ  (BQ * LQ)   // 384 independent graphs

typedef __attribute__((ext_vector_type(16))) _Float16 v16h;
typedef __attribute__((ext_vector_type(8)))  float    v8f;

__device__ __forceinline__ float sigmf(float x) { return 1.0f / (1.0f + __expf(-x)); }

// ---------------------------------------------------------------------------
// Kernel 1: embedding + LSTM. One block = 16 sequences, 16 threads/sequence,
// each thread owns 4 hidden units. Weights live in LDS (transposed so the 16
// lanes of a sequence read consecutive-by-4 addresses -> conflict free).
// h written to ws as ((b*L+l)*N+n)*64+j so the GNN kernel sees graph-major.
// ---------------------------------------------------------------------------
__global__ __launch_bounds__(256, 1) void lstm_kernel(
    const float* __restrict__ aqi,   // [B,L,N,1]
    const float* __restrict__ embW,  // [1,16]
    const float* __restrict__ embB,  // [16]
    const float* __restrict__ Wih,   // [256,16]
    const float* __restrict__ Whh,   // [256,64]
    const float* __restrict__ bih,   // [256]
    const float* __restrict__ bhh,   // [256]
    float* __restrict__ hout)        // [T*N*64]
{
  __shared__ float sWih[EMQ * 256];  // transposed: [k][row]
  __shared__ float sWhh[HQ * 256];   // transposed: [k][row]
  __shared__ float sB[256];
  __shared__ float sEw[EMQ];
  __shared__ float sEb[EMQ];
  __shared__ float sH[16][HQ];
  __shared__ float sX[16][EMQ];

  const int tid = threadIdx.x;
  for (int i = tid; i < 256 * EMQ; i += 256) sWih[(i & 15) * 256 + (i >> 4)] = Wih[i];
  for (int i = tid; i < 256 * HQ;  i += 256) sWhh[(i & 63) * 256 + (i >> 6)] = Whh[i];
  sB[tid] = bih[tid] + bhh[tid];
  if (tid < EMQ) { sEw[tid] = embW[tid]; sEb[tid] = embB[tid]; }

  const int sl  = tid >> 4;                 // sequence slot in block
  const int ln  = tid & 15;                 // lane within sequence
  const int seq = blockIdx.x * 16 + sl;     // 0..4095 = b*N + n
  const int bb  = seq >> 8;
  const int nn  = seq & 255;
  const int j0  = ln * 4;                   // hidden units owned by thread

  float cst[4] = {0.f, 0.f, 0.f, 0.f};
  #pragma unroll
  for (int jj = 0; jj < 4; ++jj) sH[sl][j0 + jj] = 0.0f;
  __syncthreads();

  for (int l = 0; l < LQ; ++l) {
    const float av = aqi[(bb * LQ + l) * NQ + nn];
    sX[sl][ln] = av * sEw[ln] + sEb[ln];
    __syncthreads();

    float hn[4];
    #pragma unroll
    for (int jj = 0; jj < 4; ++jj) {
      const int j = j0 + jj;
      float a0 = sB[j], a1 = sB[64 + j], a2 = sB[128 + j], a3 = sB[192 + j];
      #pragma unroll
      for (int k = 0; k < EMQ; ++k) {
        const float xv = sX[sl][k];
        const float* wp = &sWih[k * 256 + j];
        a0 += wp[0] * xv; a1 += wp[64] * xv; a2 += wp[128] * xv; a3 += wp[192] * xv;
      }
      #pragma unroll 8
      for (int k = 0; k < HQ; ++k) {
        const float hv = sH[sl][k];
        const float* wp = &sWhh[k * 256 + j];
        a0 += wp[0] * hv; a1 += wp[64] * hv; a2 += wp[128] * hv; a3 += wp[192] * hv;
      }
      const float ig = sigmf(a0), fg = sigmf(a1), gg = tanhf(a2), og = sigmf(a3);
      cst[jj] = fg * cst[jj] + ig * gg;
      hn[jj]  = og * tanhf(cst[jj]);
    }
    __syncthreads();   // all reads of old sH complete
    #pragma unroll
    for (int jj = 0; jj < 4; ++jj) sH[sl][j0 + jj] = hn[jj];
    float4 o4; o4.x = hn[0]; o4.y = hn[1]; o4.z = hn[2]; o4.w = hn[3];
    *(float4*)&hout[((size_t)((bb * LQ + l) * NQ + nn)) * HQ + j0] = o4;
    __syncthreads();
  }
}

// ---------------------------------------------------------------------------
// Kernel 2: fused CityGNN. One block per graph t (all 384 graphs share the
// same edge structure). Phase 1: 16-edge x 64-out WMMA tiles for mlp1
// (K = 130 padded to 160) with per-lane A fragments gathered straight from
// the L2-resident h slab, f16 weights pre-swizzled in LDS; scatter into an
// LDS accumulator via ds_add_f32 (no global atomics). Phase 2: scatter_mean
// divide + concat + mlp2 (K=128) with WMMA, streamed to d_out.
// ---------------------------------------------------------------------------
__global__ __launch_bounds__(256, 1) void gnn_kernel(
    const float* __restrict__ h,     // [T*N*64]
    const int*   __restrict__ conn,  // [2,E]
    const float* __restrict__ ew,    // [E,2]
    const float* __restrict__ w1,    // [130,64]
    const float* __restrict__ b1,    // [64]
    const float* __restrict__ w2,    // [128,64]
    const float* __restrict__ b2,    // [64]
    float* __restrict__ out)         // [T*N*64]
{
  __shared__ float sAgg[NQ * GQ];                       // 64 KB accumulator
  __shared__ __align__(32) _Float16 sW[5 * 4 * 32 * 16]; // 20 KB swizzled B frags
  __shared__ int sCnt[NQ];

  const int t    = blockIdx.x;
  const int tid  = threadIdx.x;
  const int wv   = tid >> 5;        // wave id (8 waves)
  const int lane = tid & 31;
  const int m16  = lane & 15;       // A-matrix row / D-matrix column (lane%16)
  const int hi   = lane >> 4;       // lane half select
  const int ksel = hi * 8;          // A fragment K base offset per lane half
  const float* ht  = h + (size_t)t * NQ * HQ;
  const int* rowg  = conn;
  const int* colg  = conn + EQ;

  // ---- phase 0: zero accumulators, build degree, stage swizzled mlp1_W ----
  for (int i = tid; i < NQ * GQ; i += 256) sAgg[i] = 0.0f;
  for (int i = tid; i < NQ; i += 256) sCnt[i] = 0;
  for (int i = tid; i < 5 * 4 * 32 * 16; i += 256) {
    const int p  = i & 15;
    const int l2 = (i >> 4) & 31;
    const int nt = (i >> 9) & 3;
    const int kt = i >> 11;
    const int k  = kt * 32 + ((l2 >> 4) << 3) + (p & 7) + ((p >> 3) << 4);   // B layout: lanes 16-31 hold K+8
    const int n  = nt * 16 + (l2 & 15);
    sW[i] = (_Float16)((k < 130) ? w1[k * GQ + n] : 0.0f);
  }
  __syncthreads();
  for (int e = tid; e < EQ; e += 256) atomicAdd(&sCnt[colg[e]], 1);

  float bia1[4], bia2[4];
  #pragma unroll
  for (int nt = 0; nt < 4; ++nt) { bia1[nt] = b1[nt * 16 + m16]; bia2[nt] = b2[nt * 16 + m16]; }

  // ---- phase 1: edge MLP + LDS scatter (256 tiles of 16 edges, 8 waves) ----
  for (int tile = wv; tile < EQ / 16; tile += 8) {
    const int ebase = tile * 16;
    __builtin_prefetch(&colg[(ebase + 128) & (EQ - 1)], 0, 0);
    const int e = ebase + m16;
    const int r = rowg[e];
    const int c = colg[e];
    const float* fr = ht + r * HQ;
    const float* fc = ht + c * HQ;

    v8f acc[4] = {};
    #pragma unroll
    for (int kt = 0; kt < 5; ++kt) {
      v16h a;
      if (kt < 4) {
        // two 8-element runs: [kb, kb+7] and [kb+16, kb+23]; each run lies
        // entirely inside the h[row] (k<64) or h[col] (k>=64) segment.
        const int kb = kt * 32 + ksel;
        const float* s1 = (kb      < 64) ? (fr + kb)       : (fc + (kb - 64));
        const float* s2 = (kb + 16 < 64) ? (fr + kb + 16)  : (fc + (kb - 48));
        const float4 r0 = ((const float4*)s1)[0];
        const float4 r1 = ((const float4*)s1)[1];
        const float4 r2 = ((const float4*)s2)[0];
        const float4 r3 = ((const float4*)s2)[1];
        a[0]=(_Float16)r0.x; a[1]=(_Float16)r0.y; a[2]=(_Float16)r0.z; a[3]=(_Float16)r0.w;
        a[4]=(_Float16)r1.x; a[5]=(_Float16)r1.y; a[6]=(_Float16)r1.z; a[7]=(_Float16)r1.w;
        a[8]=(_Float16)r2.x; a[9]=(_Float16)r2.y; a[10]=(_Float16)r2.z; a[11]=(_Float16)r2.w;
        a[12]=(_Float16)r3.x; a[13]=(_Float16)r3.y; a[14]=(_Float16)r3.z; a[15]=(_Float16)r3.w;
      } else {
        #pragma unroll
        for (int p = 0; p < 16; ++p) a[p] = (_Float16)0.0f;
        if (hi == 0) { a[0] = (_Float16)ew[e * 2 + 0]; a[1] = (_Float16)ew[e * 2 + 1]; }
      }
      #pragma unroll
      for (int nt = 0; nt < 4; ++nt) {
        const v16h bf = *(const v16h*)&sW[(((kt * 4 + nt) * 32 + lane) << 4)];
        acc[nt] = __builtin_amdgcn_wmma_f32_16x16x32_f16(
            false, a, false, bf, (short)0, acc[nt], false, false);
      }
    }
    // D element (m = v + 8*hi, n = nt*16 + m16): add bias, scatter into LDS agg
    const int mb = hi * 8;
    #pragma unroll
    for (int v = 0; v < 8; ++v) {
      const int cn = colg[ebase + mb + v];
      #pragma unroll
      for (int nt = 0; nt < 4; ++nt) {
        atomicAdd(&sAgg[cn * GQ + nt * 16 + m16], acc[nt][v] + bia1[nt]);
      }
    }
  }
  __syncthreads();

  // ---- stage swizzled mlp2_W (K=128 -> 4 K-tiles) ----
  for (int i = tid; i < 4 * 4 * 32 * 16; i += 256) {
    const int p  = i & 15;
    const int l2 = (i >> 4) & 31;
    const int nt = (i >> 9) & 3;
    const int kt = i >> 11;
    const int k  = kt * 32 + ((l2 >> 4) << 3) + (p & 7) + ((p >> 3) << 4);
    const int n  = nt * 16 + (l2 & 15);
    sW[i] = (_Float16)w2[k * GQ + n];
  }
  __syncthreads();

  // ---- phase 2: concat(h, agg/deg) @ mlp2_W, 16 node tiles ----
  for (int tile = wv; tile < 16; tile += 8) {
    const int node = tile * 16 + m16;
    const float* fh = ht + node * HQ;
    const float rcv = 1.0f / fmaxf((float)sCnt[node], 1.0f);

    v8f acc[4] = {};
    #pragma unroll
    for (int kt = 0; kt < 4; ++kt) {
      const int kb = kt * 32 + ksel;
      v16h a;
      if (kt < 2) {  // k < 64: LSTM hidden
        const float4 r0 = ((const float4*)(fh + kb))[0];
        const float4 r1 = ((const float4*)(fh + kb))[1];
        const float4 r2 = ((const float4*)(fh + kb + 16))[0];
        const float4 r3 = ((const float4*)(fh + kb + 16))[1];
        a[0]=(_Float16)r0.x; a[1]=(_Float16)r0.y; a[2]=(_Float16)r0.z; a[3]=(_Float16)r0.w;
        a[4]=(_Float16)r1.x; a[5]=(_Float16)r1.y; a[6]=(_Float16)r1.z; a[7]=(_Float16)r1.w;
        a[8]=(_Float16)r2.x; a[9]=(_Float16)r2.y; a[10]=(_Float16)r2.z; a[11]=(_Float16)r2.w;
        a[12]=(_Float16)r3.x; a[13]=(_Float16)r3.y; a[14]=(_Float16)r3.z; a[15]=(_Float16)r3.w;
      } else {       // k >= 64: scatter_mean aggregate from LDS
        const float* ag = &sAgg[node * GQ + (kb - 64)];
        #pragma unroll
        for (int p = 0; p < 8; ++p) {
          a[p]     = (_Float16)(ag[p]      * rcv);
          a[p + 8] = (_Float16)(ag[p + 16] * rcv);
        }
      }
      #pragma unroll
      for (int nt = 0; nt < 4; ++nt) {
        const v16h bf = *(const v16h*)&sW[(((kt * 4 + nt) * 32 + lane) << 4)];
        acc[nt] = __builtin_amdgcn_wmma_f32_16x16x32_f16(
            false, a, false, bf, (short)0, acc[nt], false, false);
      }
    }
    const int mb = hi * 8;
    #pragma unroll
    for (int v = 0; v < 8; ++v) {
      const int nd = tile * 16 + mb + v;
      float* op = out + ((size_t)t * NQ + nd) * GQ;
      #pragma unroll
      for (int nt = 0; nt < 4; ++nt) op[nt * 16 + m16] = acc[nt][v] + bia2[nt];
    }
  }
}

extern "C" void kernel_launch(void* const* d_in, const int* in_sizes, int n_in,
                              void* d_out, int out_size, void* d_ws, size_t ws_size,
                              hipStream_t stream) {
  (void)in_sizes; (void)n_in; (void)out_size; (void)ws_size;
  const float* aqi  = (const float*)d_in[0];
  const int*   conn = (const int*)d_in[1];
  const float* ew   = (const float*)d_in[2];
  const float* embW = (const float*)d_in[3];
  const float* embB = (const float*)d_in[4];
  const float* Wih  = (const float*)d_in[5];
  const float* Whh  = (const float*)d_in[6];
  const float* bih  = (const float*)d_in[7];
  const float* bhh  = (const float*)d_in[8];
  const float* w1   = (const float*)d_in[9];
  const float* b1   = (const float*)d_in[10];
  const float* w2   = (const float*)d_in[11];
  const float* b2   = (const float*)d_in[12];
  float* hbuf = (float*)d_ws;        // 384*256*64 f32 = 25.2 MB
  float* outp = (float*)d_out;

  lstm_kernel<<<(BQ * NQ) / 16, 256, 0, stream>>>(aqi, embW, embB, Wih, Whh, bih, bhh, hbuf);
  gnn_kernel<<<TQ, 256, 0, stream>>>(hbuf, conn, ew, w1, b1, w2, b2, outp);
}
